// BayesianNN_58884001628320
// MI455X (gfx1250) — compile-verified, matrix operand order checked by reading, not run on hardware
//
#include <hip/hip_runtime.h>
#include <hip/hip_bf16.h>
#include <math.h>

// ---------------------------------------------------------------------------
// Problem constants (from reference)
// ---------------------------------------------------------------------------
#define D_DIM 7686
#define M_DIM 128
#define N_HID 128
#define N_OUTS 2
#define N_DIM (D_DIM + N_HID + N_OUTS)   // 7816
#define NNI   (N_HID + N_OUTS)           // 130 non-input nodes

#define KSTEP 32
#define KFULL (D_DIM / KSTEP)            // 240 full bf16 k-steps
#define NTILES ((D_DIM + 15) / 16)       // 481 column tiles of 16

typedef __attribute__((ext_vector_type(16))) __bf16 v16bf;
typedef __attribute__((ext_vector_type(8)))  float  v8f;

// ---------------------------------------------------------------------------
// Fragment loaders: fp32 row -> bf16 A/B fragment for v_wmma_f32_16x16x32_bf16
// Per ISA layout (16-bit 16x32 A): lane L holds M=L%16; kh=L/16 selects the
// 8-wide K half. Element i<8 -> K = k0+kh*8+i ; i>=8 -> K = k0+16+kh*8+(i-8).
// kbase passed in is (k0 + kh*8); runs are 8-byte aligned (D even, k0%32==0).
// ---------------------------------------------------------------------------
__device__ inline v16bf load_frag_full(const float* __restrict__ row, int kbase) {
  v16bf f;
  const float2* p0 = reinterpret_cast<const float2*>(row + kbase);
  const float2* p1 = reinterpret_cast<const float2*>(row + kbase + 16);
#pragma unroll
  for (int i = 0; i < 4; ++i) {
    float2 a = p0[i];
    float2 b = p1[i];
    f[2*i]     = (__bf16)a.x;  f[2*i + 1] = (__bf16)a.y;
    f[8 + 2*i] = (__bf16)b.x;  f[9 + 2*i] = (__bf16)b.y;
  }
  return f;
}

__device__ inline v16bf load_frag_tail(const float* __restrict__ row, int k0, int kh) {
  v16bf f;
#pragma unroll
  for (int i = 0; i < 16; ++i) {
    int k = (i < 8) ? (k0 + kh*8 + i) : (k0 + 16 + kh*8 + (i - 8));
    f[i] = (k < D_DIM) ? (__bf16)row[k] : (__bf16)0.0f;
  }
  return f;
}

// ---------------------------------------------------------------------------
// QKV projection: out[m,n] = sum_k x[m,k]*W[n,k] + bias[n]
// One wave owns a full 128x16 output stripe (8 accumulators) so each weight
// row is streamed from HBM exactly once; x (3.9 MB) stays hot in L2.
// All 8 A fragments + B fragment are materialized in disjoint registers
// BEFORE the WMMA burst, so the 8 back-to-back WMMAs have no WAR hazard on
// their A/B operands (avoids the 4-coexec-NOP penalty per WMMA) and the
// loads form one big clause (more MLP for the HBM-bound stream).
// grid = (ceil(481/4), 3), block = 128 (4 waves).
// ---------------------------------------------------------------------------
__global__ __launch_bounds__(128)
void qkv_gemm_wmma(const float* __restrict__ x,
                   const float* __restrict__ Wq, const float* __restrict__ bq,
                   const float* __restrict__ Wk, const float* __restrict__ bk,
                   const float* __restrict__ Wv, const float* __restrict__ bv,
                   float* __restrict__ qo, float* __restrict__ ko,
                   float* __restrict__ vo)
{
  const int wave = threadIdx.x >> 5;
  const int lane = threadIdx.x & 31;
  const int ntile = blockIdx.x * 4 + wave;
  if (ntile >= NTILES) return;

  const float* W; const float* bias; float* out;
  if (blockIdx.y == 0)      { W = Wq; bias = bq; out = qo; }
  else if (blockIdx.y == 1) { W = Wk; bias = bk; out = ko; }
  else                      { W = Wv; bias = bv; out = vo; }

  const int nl = lane & 15;       // A: local M ; B: local N ; C/D: local N
  const int kh = lane >> 4;       // which 8-wide K half this lane covers
  const int n = ntile * 16 + nl;
  const int nrow = (n < D_DIM) ? n : (D_DIM - 1);   // clamp; invalid cols never stored
  const float* wrow = W + (size_t)nrow * D_DIM;
  const float* xrow = x + (size_t)nl * D_DIM;       // m-tile stride added via offset

  v8f acc[8];
#pragma unroll
  for (int t = 0; t < 8; ++t)
#pragma unroll
    for (int e = 0; e < 8; ++e) acc[t][e] = 0.0f;

  for (int ks = 0; ks < KFULL; ++ks) {
    const int kbase = ks * KSTEP + kh * 8;
    __builtin_prefetch(wrow + kbase + 2 * KSTEP, 0, 0);   // global_prefetch_b8

    // Stage everything first (disjoint registers -> no WMMA A/B WAR hazard).
    v16bf bfrag = load_frag_full(wrow, kbase);
    v16bf af0 = load_frag_full(xrow + (size_t)(0 * 16) * D_DIM, kbase);
    v16bf af1 = load_frag_full(xrow + (size_t)(1 * 16) * D_DIM, kbase);
    v16bf af2 = load_frag_full(xrow + (size_t)(2 * 16) * D_DIM, kbase);
    v16bf af3 = load_frag_full(xrow + (size_t)(3 * 16) * D_DIM, kbase);
    v16bf af4 = load_frag_full(xrow + (size_t)(4 * 16) * D_DIM, kbase);
    v16bf af5 = load_frag_full(xrow + (size_t)(5 * 16) * D_DIM, kbase);
    v16bf af6 = load_frag_full(xrow + (size_t)(6 * 16) * D_DIM, kbase);
    v16bf af7 = load_frag_full(xrow + (size_t)(7 * 16) * D_DIM, kbase);

    // 8 independent back-to-back WMMAs (disjoint D, shared B, disjoint A).
    acc[0] = __builtin_amdgcn_wmma_f32_16x16x32_bf16(false, af0, false, bfrag, (short)0, acc[0], false, false);
    acc[1] = __builtin_amdgcn_wmma_f32_16x16x32_bf16(false, af1, false, bfrag, (short)0, acc[1], false, false);
    acc[2] = __builtin_amdgcn_wmma_f32_16x16x32_bf16(false, af2, false, bfrag, (short)0, acc[2], false, false);
    acc[3] = __builtin_amdgcn_wmma_f32_16x16x32_bf16(false, af3, false, bfrag, (short)0, acc[3], false, false);
    acc[4] = __builtin_amdgcn_wmma_f32_16x16x32_bf16(false, af4, false, bfrag, (short)0, acc[4], false, false);
    acc[5] = __builtin_amdgcn_wmma_f32_16x16x32_bf16(false, af5, false, bfrag, (short)0, acc[5], false, false);
    acc[6] = __builtin_amdgcn_wmma_f32_16x16x32_bf16(false, af6, false, bfrag, (short)0, acc[6], false, false);
    acc[7] = __builtin_amdgcn_wmma_f32_16x16x32_bf16(false, af7, false, bfrag, (short)0, acc[7], false, false);
  }
  { // ragged K tail (6 valid, zero-filled)
    const int k0 = KFULL * KSTEP;
    v16bf bfrag = load_frag_tail(wrow, k0, kh);
#pragma unroll
    for (int mt = 0; mt < 8; ++mt) {
      v16bf afrag = load_frag_tail(xrow + (size_t)(mt * 16) * D_DIM, k0, kh);
      acc[mt] = __builtin_amdgcn_wmma_f32_16x16x32_bf16(
          false, afrag, false, bfrag, (short)0, acc[mt], false, false);
    }
  }

  if (n < D_DIM) {
    const float bn = bias[n];
    // C/D layout: VGPR r, lanes 0-15 -> M=r ; lanes 16-31 -> M=r+8 ; N=lane%16
#pragma unroll
    for (int mt = 0; mt < 8; ++mt)
#pragma unroll
      for (int r = 0; r < 8; ++r) {
        const int m = mt * 16 + r + 8 * kh;
        out[(size_t)m * D_DIM + n] = acc[mt][r] + bn;
      }
  }
}

// ---------------------------------------------------------------------------
// Attention: per query row i, scores -> softmax -> ctx row. grid=128, block=128.
// ---------------------------------------------------------------------------
__global__ __launch_bounds__(128)
void attn_ctx(const float* __restrict__ q, const float* __restrict__ kmat,
              const float* __restrict__ v, float* __restrict__ ctx)
{
  __shared__ float sh_p[M_DIM];
  __shared__ float red[M_DIM];
  const int i = blockIdx.x;
  const int j = threadIdx.x;

  const float* qi = q + (size_t)i * D_DIM;
  const float* kj = kmat + (size_t)j * D_DIM;
  float acc = 0.0f;
#pragma unroll 4
  for (int c = 0; c < D_DIM; ++c) acc += qi[c] * kj[c];
  const float s = acc * rsqrtf((float)D_DIM);

  red[j] = s; __syncthreads();
  for (int off = 64; off > 0; off >>= 1) {
    if (j < off) red[j] = fmaxf(red[j], red[j + off]);
    __syncthreads();
  }
  const float mx = red[0]; __syncthreads();
  const float p = expf(s - mx);
  sh_p[j] = p; red[j] = p; __syncthreads();
  for (int off = 64; off > 0; off >>= 1) {
    if (j < off) red[j] += red[j + off];
    __syncthreads();
  }
  const float inv = 1.0f / red[0];
  __syncthreads();

  for (int c = j; c < D_DIM; c += M_DIM) {
    float a = 0.0f;
#pragma unroll 4
    for (int jj = 0; jj < M_DIM; ++jj) a += sh_p[jj] * v[(size_t)jj * D_DIM + c];
    ctx[(size_t)i * D_DIM + c] = a * inv;
  }
}

// pooled[c] = mean_i ctx[i,c]  (written into vals[0..D))
__global__ void pool_mean(const float* __restrict__ ctx, float* __restrict__ vals)
{
  const int c = blockIdx.x * blockDim.x + threadIdx.x;
  if (c >= D_DIM) return;
  float s = 0.0f;
  for (int i = 0; i < M_DIM; ++i) s += ctx[(size_t)i * D_DIM + c];
  vals[c] = s * (1.0f / M_DIM);
}

__global__ void zero_pre(float* __restrict__ pre) { pre[threadIdx.x] = 0.0f; }

// ---------------------------------------------------------------------------
// Pass 1: only the 130 non-input columns matter, and only k<D contributes
// (vals[k>=D]==0). W = mu + sigma*eps fused -> ~12 MB touched, not 732 MB.
// grid=64 k-chunks, block=256 (threads 0..129 own a column), atomic combine.
// ---------------------------------------------------------------------------
__global__ __launch_bounds__(256)
void hidden_partial(const float* __restrict__ vals, const float* __restrict__ Wmu,
                    const float* __restrict__ Wsig, const float* __restrict__ eps,
                    float* __restrict__ pre)
{
  const int t = threadIdx.x;
  if (t >= NNI) return;
  const int kchunk = (D_DIM + gridDim.x - 1) / gridDim.x;
  const int k0 = blockIdx.x * kchunk;
  const int k1 = (k0 + kchunk < D_DIM) ? (k0 + kchunk) : D_DIM;
  float acc = 0.0f;
  for (int kk = k0; kk < k1; ++kk) {
    const float vv = vals[kk];
    const size_t idx = (size_t)kk * N_DIM + D_DIM + t;   // contiguous across t
    acc += vv * (Wmu[idx] + Wsig[idx] * eps[idx]);
  }
  atomicAdd(&pre[t], acc);
}

__global__ void hidden_finalize(const float* __restrict__ pre,
                                const float* __restrict__ bmu,
                                const float* __restrict__ bsig,
                                const float* __restrict__ epsb,
                                float* __restrict__ vals)
{
  const int t = threadIdx.x;
  if (t >= NNI) return;
  const int n = D_DIM + t;
  const float b = bmu[n] + bsig[n] * epsb[n];
  vals[n] = tanhf(pre[t] + b);
}

// Pass 2: only the 2 output columns are observable. One block, LDS reduce.
__global__ __launch_bounds__(256)
void output_kernel(const float* __restrict__ vals, const float* __restrict__ Wmu,
                   const float* __restrict__ Wsig, const float* __restrict__ eps,
                   const float* __restrict__ bmu, const float* __restrict__ bsig,
                   const float* __restrict__ epsb, float* __restrict__ out)
{
  __shared__ float r0[256];
  __shared__ float r1[256];
  const int t = threadIdx.x;
  float a0 = 0.0f, a1 = 0.0f;
  for (int kk = t; kk < N_DIM; kk += 256) {
    const float vv = vals[kk];
    const size_t base = (size_t)kk * N_DIM + (N_DIM - 2);
    a0 += vv * (Wmu[base]     + Wsig[base]     * eps[base]);
    a1 += vv * (Wmu[base + 1] + Wsig[base + 1] * eps[base + 1]);
  }
  r0[t] = a0; r1[t] = a1; __syncthreads();
  for (int off = 128; off > 0; off >>= 1) {
    if (t < off) { r0[t] += r0[t + off]; r1[t] += r1[t + off]; }
    __syncthreads();
  }
  if (t == 0) {
    const int n0 = N_DIM - 2;
    out[0] = tanhf(r0[0] + bmu[n0]     + bsig[n0]     * epsb[n0]);
    out[1] = tanhf(r1[0] + bmu[n0 + 1] + bsig[n0 + 1] * epsb[n0 + 1]);
  }
}

// ---------------------------------------------------------------------------
extern "C" void kernel_launch(void* const* d_in, const int* in_sizes, int n_in,
                              void* d_out, int out_size, void* d_ws, size_t ws_size,
                              hipStream_t stream)
{
  (void)in_sizes; (void)n_in; (void)out_size; (void)ws_size;
  const float* x    = (const float*)d_in[0];
  const float* Wq   = (const float*)d_in[1];
  const float* bq   = (const float*)d_in[2];
  const float* Wk   = (const float*)d_in[3];
  const float* bk   = (const float*)d_in[4];
  const float* Wv   = (const float*)d_in[5];
  const float* bv   = (const float*)d_in[6];
  const float* Wmu  = (const float*)d_in[7];
  const float* Wsig = (const float*)d_in[8];
  const float* epsw = (const float*)d_in[9];
  const float* bmu  = (const float*)d_in[10];
  const float* bsig = (const float*)d_in[11];
  const float* epsb = (const float*)d_in[12];
  float* out = (float*)d_out;

  float* ws = (float*)d_ws;
  const size_t MD = (size_t)M_DIM * D_DIM;
  float* q    = ws;            // [M,D]
  float* k    = q + MD;        // [M,D]
  float* v    = k + MD;        // [M,D]
  float* ctx  = v + MD;        // [M,D]
  float* vals = ctx + MD;      // [N]
  float* pre  = vals + ((N_DIM + 255) / 256) * 256;  // [130] (padded)

  dim3 gemm_grid((NTILES + 3) / 4, 3, 1);
  qkv_gemm_wmma<<<gemm_grid, 128, 0, stream>>>(x, Wq, bq, Wk, bk, Wv, bv, q, k, v);
  attn_ctx<<<dim3(M_DIM), 128, 0, stream>>>(q, k, v, ctx);
  pool_mean<<<dim3((D_DIM + 255) / 256), 256, 0, stream>>>(ctx, vals);
  zero_pre<<<1, 256, 0, stream>>>(pre);
  hidden_partial<<<dim3(64), 256, 0, stream>>>(vals, Wmu, Wsig, epsw, pre);
  hidden_finalize<<<1, 160, 0, stream>>>(pre, bmu, bsig, epsb, vals);
  output_kernel<<<1, 256, 0, stream>>>(vals, Wmu, Wsig, epsw, bmu, bsig, epsb, out);
}
